// FlashLTCLayer_24043226923373
// MI455X (gfx1250) — compile-verified, matrix operand order checked by reading, not run on hardware
//
#include <hip/hip_runtime.h>
#include <hip/hip_bf16.h>
#include <stdint.h>

// Problem dims (fixed by reference)
#define B_    32
#define L_    1024
#define D_    512
#define H_    512
#define NSTEP 6
#define NWG   4              // persistent workgroups in the scan
#define HC    (H_ / NWG)     // 128 output columns per workgroup
#define BL    (B_ * L_)      // 32768

typedef __attribute__((ext_vector_type(16))) __bf16 v16bf;
typedef __attribute__((ext_vector_type(8)))  __bf16 v8bf;   // 16 bytes
typedef __attribute__((ext_vector_type(8)))  float  v8f;
typedef __attribute__((ext_vector_type(4)))  unsigned int v4u;
typedef __attribute__((ext_vector_type(8)))  int v8i;
typedef __attribute__((ext_vector_type(4)))  int v4i;

#define WMMA_BF16(a, b, c) \
  __builtin_amdgcn_wmma_f32_16x16x32_bf16(false, (a), false, (b), (short)0, (c), false, false)

// Load one 16x32 bf16 operand fragment from a row-major matrix.
// ISA 7.12.2: lane = half*16 + row; VGPR0-3 hold K = 8*half + 0..7,
// VGPR4-7 hold K = 16 + 8*half + 0..7  -> two contiguous 16-byte chunks.
// Works for both A (rows of the left matrix) and B (rows of W == cols of W^T).
static __device__ __forceinline__ v16bf load_frag(const __bf16* base, int ld, int k) {
  const int lane = threadIdx.x & 31;
  const int row  = lane & 15;
  const int half = lane >> 4;
  const __bf16* p = base + row * ld + k + 8 * half;
  v8bf lo = *(const v8bf*)(p);
  v8bf hi = *(const v8bf*)(p + 16);
  v16bf r;
#pragma unroll
  for (int i = 0; i < 8; ++i) { r[i] = lo[i]; r[i + 8] = hi[i]; }
  return r;
}

// ------------------------------------------------------------ TDM staging
// DMA one 128-row x 1024-byte weight panel (128 KB) global -> LDS using the
// Tensor Data Mover (ISA ch. 8). 2D descriptor, data_size = 8 bytes:
// tensor/tile = 128 x 128 elements, dim0 stride = 128 elements.
// Tracked by TENSORcnt; issued by one wave, completion fanned out via barrier.
// This toolchain's builtin takes 6 args: (g0 u32x4, g1 i32x8, g2 i32x4,
// g3 i32x4, g4 i32x8, cpol i32).
static __device__ __forceinline__ void tdm_load_panel(const __bf16* gsrc, __bf16* ldst) {
  const unsigned lds_addr = (unsigned)(uintptr_t)ldst;      // LDS offset = low 32 bits
  const unsigned long long ga = (unsigned long long)(uintptr_t)gsrc;
  // group0: [1:0]=count=1 | [63:32]=lds_addr | [120:64]=global_addr | [127:126]=type 2
  v4u g0 = { 1u,
             lds_addr,
             (unsigned)ga,
             (unsigned)((ga >> 32) & 0x01FFFFFFu) | 0x80000000u };
  // group1: data_size=3 (8B) | tensor_dim0=128 | tensor_dim1=128 |
  //         tile_dim0=128 | tile_dim1=128 | tensor_dim0_stride=128
  v8i g1 = { (int)(3u << 16),
             (int)(128u << 16),     // tensor_dim0[15:0] at bits 63:48
             (int)(128u << 16),     // tensor_dim0[31:16]=0, tensor_dim1[15:0]
             (int)(128u << 16),     // tensor_dim1[31:16]=0, tile_dim0
             128,                   // tile_dim1 (tile_dim2 = 0)
             128,                   // tensor_dim0_stride[31:0]
             0, 0 };
  v4i gz4 = { 0, 0, 0, 0 };         // groups 2/3: dims 2..4 unused
  v8i gz8 = { 0, 0, 0, 0, 0, 0, 0, 0 };
  __builtin_amdgcn_tensor_load_to_lds(g0, g1, gz4, gz4, gz8, 0);
}

// ---------------------------------------------------------------- conversion
__global__ void k_zero_sync(unsigned int* cnt) {
  if (threadIdx.x < 64) cnt[threadIdx.x] = 0u;
}

__global__ void k_convert_x(const float* __restrict__ x, __bf16* __restrict__ xb, int n) {
  int i = blockIdx.x * blockDim.x + threadIdx.x;
  if (i < n) xb[i] = (__bf16)x[i];
}

__global__ void k_pack_weights(const float* __restrict__ W_in, const float* __restrict__ W_rec,
                               const float* __restrict__ W_gate, const float* __restrict__ W_out,
                               const float* __restrict__ log_tau,
                               __bf16* __restrict__ Winb, __bf16* __restrict__ Wrecb,
                               __bf16* __restrict__ Wgxb, __bf16* __restrict__ Wghb,
                               __bf16* __restrict__ Woutb, float* __restrict__ rate) {
  int i = blockIdx.x * blockDim.x + threadIdx.x;
  if (i >= H_ * D_) return;
  Winb[i]  = (__bf16)W_in[i];
  Wrecb[i] = (__bf16)W_rec[i];
  Woutb[i] = (__bf16)W_out[i];
  int r = i >> 9, c = i & (D_ - 1);
  Wgxb[i] = (__bf16)W_gate[r * (D_ + H_) + c];        // W_gate[:, :D]
  Wghb[i] = (__bf16)W_gate[r * (D_ + H_) + D_ + c];   // W_gate[:, D:]
  if (i < H_) rate[i] = (1.0f / NSTEP) * __expf(-log_tau[i]);  // sub_dt / tau
}

// ------------------------------------------------- fused input GEMM (drive, gate_x)
// block = 8 waves; block owns a 16-row M strip; wave w owns N columns [64w, 64w+64)
// for BOTH outputs (shares the A fragment). Software-pipelined: while the 8 WMMAs
// of chunk k execute, the A + 8 B fragments of chunk k+32 are fetched into a second
// register set (prefetch index wraps so there is no branch and no OOB access).
// Stores transposed to (L,B,H) so the scan's per-timestep reads are contiguous.
__global__ void __launch_bounds__(256) k_input_gemm(const __bf16* __restrict__ xb,
                                                    const __bf16* __restrict__ Winb,
                                                    const __bf16* __restrict__ Wgxb,
                                                    const float* __restrict__ b_in,
                                                    const float* __restrict__ b_gate,
                                                    float* __restrict__ drive_t,
                                                    float* __restrict__ gate_t) {
  const int blk  = blockIdx.x;        // 16-row tile of the (BL, D) matrix
  const int wave = threadIdx.x >> 5;
  const int n0   = wave * 64;
  const __bf16* Abase = xb + (size_t)blk * 16 * D_;
  v8f accD[4] = {};
  v8f accG[4] = {};

  v16bf a_c = load_frag(Abase, D_, 0);
  v16bf bD_c[4], bG_c[4];
#pragma unroll
  for (int j = 0; j < 4; ++j) {
    bD_c[j] = load_frag(Winb + (size_t)(n0 + 16 * j) * D_, D_, 0);
    bG_c[j] = load_frag(Wgxb + (size_t)(n0 + 16 * j) * D_, D_, 0);
  }
#pragma unroll
  for (int k = 0; k < D_; k += 32) {
    const int kn = (k + 32) & (D_ - 1);         // wrap-around prefetch (last discarded)
    v16bf a_n = load_frag(Abase, D_, kn);
    v16bf bD_n[4], bG_n[4];
#pragma unroll
    for (int j = 0; j < 4; ++j) {
      bD_n[j] = load_frag(Winb + (size_t)(n0 + 16 * j) * D_, D_, kn);
      bG_n[j] = load_frag(Wgxb + (size_t)(n0 + 16 * j) * D_, D_, kn);
    }
#pragma unroll
    for (int j = 0; j < 4; ++j) {
      accD[j] = WMMA_BF16(a_c, bD_c[j], accD[j]);
      accG[j] = WMMA_BF16(a_c, bG_c[j], accG[j]);
    }
    a_c = a_n;
#pragma unroll
    for (int j = 0; j < 4; ++j) { bD_c[j] = bD_n[j]; bG_c[j] = bG_n[j]; }
  }

  const int lane = threadIdx.x & 31;
  const int ncol = lane & 15, half = lane >> 4;
#pragma unroll
  for (int j = 0; j < 4; ++j) {
    int col = n0 + 16 * j + ncol;
    float bi = b_in[col], bg = b_gate[col];
#pragma unroll
    for (int v = 0; v < 8; ++v) {
      int m  = v + 8 * half;
      int bl = blk * 16 + m;
      int b  = bl >> 10, l = bl & (L_ - 1);
      size_t idx = ((size_t)l * B_ + b) * H_ + col;   // (L, B, H)
      drive_t[idx] = accD[j][v] + bi;
      gate_t[idx]  = accG[j][v] + bg;
    }
  }
}

// ------------------------------------------------------------ output GEMM
__global__ void __launch_bounds__(256) k_out_gemm(const __bf16* __restrict__ fwd_bf,
                                                  const __bf16* __restrict__ Woutb,
                                                  const float* __restrict__ b_out,
                                                  float* __restrict__ out) {
  const int blk  = blockIdx.x;
  const int wave = threadIdx.x >> 5;
  const int n0   = wave * 64;
  const __bf16* Abase = fwd_bf + (size_t)blk * 16 * H_;
  v8f acc[4] = {};

  v16bf a_c = load_frag(Abase, H_, 0);
  v16bf b_c[4];
#pragma unroll
  for (int j = 0; j < 4; ++j)
    b_c[j] = load_frag(Woutb + (size_t)(n0 + 16 * j) * H_, H_, 0);
#pragma unroll
  for (int k = 0; k < H_; k += 32) {
    const int kn = (k + 32) & (H_ - 1);
    v16bf a_n = load_frag(Abase, H_, kn);
    v16bf b_n[4];
#pragma unroll
    for (int j = 0; j < 4; ++j)
      b_n[j] = load_frag(Woutb + (size_t)(n0 + 16 * j) * H_, H_, kn);
#pragma unroll
    for (int j = 0; j < 4; ++j)
      acc[j] = WMMA_BF16(a_c, b_c[j], acc[j]);
    a_c = a_n;
#pragma unroll
    for (int j = 0; j < 4; ++j) b_c[j] = b_n[j];
  }

  const int lane = threadIdx.x & 31;
  const int ncol = lane & 15, half = lane >> 4;
#pragma unroll
  for (int j = 0; j < 4; ++j) {
    int col = n0 + 16 * j + ncol;
    float bo = b_out[col];
#pragma unroll
    for (int v = 0; v < 8; ++v) {
      int m = v + 8 * half;
      out[((size_t)blk * 16 + m) * H_ + col] = acc[j][v] + bo;
    }
  }
}

// ------------------------------------------------------------------ scan
// Cross-WG sequence barrier through L2, then refresh the full h panel in LDS.
static __device__ __forceinline__ void sync_and_load_h(unsigned int* cnt, unsigned int target,
                                                       __bf16* sH, const __bf16* hg) {
  __threadfence();           // make this WG's h stores visible at device scope
  __syncthreads();           // all waves done writing h AND done reading old sH
  if (threadIdx.x == 0) {
    __hip_atomic_fetch_add(cnt, 1u, __ATOMIC_RELEASE, __HIP_MEMORY_SCOPE_AGENT);
    while (__hip_atomic_load(cnt, __ATOMIC_ACQUIRE, __HIP_MEMORY_SCOPE_AGENT) < target)
      __builtin_amdgcn_s_sleep(1);
  }
  __syncthreads();
  const uint4* src = (const uint4*)hg;       // 32 KB: full h, all panels
  uint4* dst = (uint4*)sH;
  for (int i = threadIdx.x; i < (B_ * H_ * 2 / 16); i += blockDim.x) dst[i] = src[i];
  __syncthreads();
}

// 4 persistent WGs; WG p owns output columns [p*128, p*128+128).
// LDS: W_rec panel (128KB bf16) + Wg_h panel (128KB) + full h (32KB) = 288KB < 320KB.
// Weight panels staged by the Tensor Data Mover (one descriptor each, wave 0).
// Wave w owns 16 columns; computes both 16-row M tiles for both GEMMs (4 WMMAs/k-chunk).
__global__ void __launch_bounds__(256, 1) k_scan(const float* __restrict__ drive_t,
                                                 const float* __restrict__ gate_t,
                                                 const __bf16* __restrict__ Wrecb,
                                                 const __bf16* __restrict__ Wghb,
                                                 const float* __restrict__ rate,
                                                 __bf16* __restrict__ fwd_bf,
                                                 __bf16* hg, unsigned int* cnt,
                                                 float* __restrict__ h_last) {
  extern __shared__ char smem[];
  __bf16* sWrec = (__bf16*)smem;                         // HC x H_
  __bf16* sWgh  = sWrec + (size_t)HC * H_;               // HC x H_
  __bf16* sH    = sWgh  + (size_t)HC * H_;               // B_ x H_

  const int part  = blockIdx.x;
  const int wave  = threadIdx.x >> 5;
  const int lane  = threadIdx.x & 31;
  const int ncol  = lane & 15;
  const int halfm = lane >> 4;
  const int nloc0 = wave * 16;                 // local column base in panel
  const int gcol  = part * HC + nloc0 + ncol;  // global h column

  // Stage both weight panels into LDS via TDM (reused for 6144 substeps).
  if (threadIdx.x < 32) {                      // wave 0 issues both DMAs
    tdm_load_panel(Wrecb + (size_t)part * HC * H_, sWrec);
    tdm_load_panel(Wghb  + (size_t)part * HC * H_, sWgh);
    __builtin_amdgcn_s_wait_tensorcnt(0);
  }
  __syncthreads();

  float h[2][8];
#pragma unroll
  for (int mt = 0; mt < 2; ++mt)
#pragma unroll
    for (int v = 0; v < 8; ++v) h[mt][v] = 0.f;

  // publish initial h = 0 for our columns
#pragma unroll
  for (int mt = 0; mt < 2; ++mt)
#pragma unroll
    for (int v = 0; v < 8; ++v)
      hg[(size_t)(16 * mt + v + 8 * halfm) * H_ + gcol] = (__bf16)0.f;

  const float rt = rate[gcol];

  unsigned int target = NWG;
  sync_and_load_h(cnt, target, sH, hg);

  for (int l = 0; l < L_; ++l) {
    float dreg[2][8], gxreg[2][8];
#pragma unroll
    for (int mt = 0; mt < 2; ++mt)
#pragma unroll
      for (int v = 0; v < 8; ++v) {
        int b = 16 * mt + v + 8 * halfm;
        size_t idx = ((size_t)l * B_ + b) * H_ + gcol;
        dreg[mt][v]  = drive_t[idx];
        gxreg[mt][v] = gate_t[idx];
      }
    for (int s = 0; s < NSTEP; ++s) {
      v8f accR[2] = {};
      v8f accG[2] = {};
      for (int k = 0; k < H_; k += 32) {
        v16bf bR = load_frag(sWrec + (size_t)nloc0 * H_, H_, k);
        v16bf bG = load_frag(sWgh  + (size_t)nloc0 * H_, H_, k);
        v16bf a0 = load_frag(sH, H_, k);
        v16bf a1 = load_frag(sH + (size_t)16 * H_, H_, k);
        accR[0] = WMMA_BF16(a0, bR, accR[0]);
        accG[0] = WMMA_BF16(a0, bG, accG[0]);
        accR[1] = WMMA_BF16(a1, bR, accR[1]);
        accG[1] = WMMA_BF16(a1, bG, accG[1]);
      }
#pragma unroll
      for (int mt = 0; mt < 2; ++mt)
#pragma unroll
        for (int v = 0; v < 8; ++v) {
          float g = 1.f / (1.f + __expf(-(gxreg[mt][v] + accG[mt][v])));
          float t = tanhf(dreg[mt][v] + accR[mt][v]);
          float f = -h[mt][v] + g * t;
          h[mt][v] += rt * f;
        }
#pragma unroll
      for (int mt = 0; mt < 2; ++mt)
#pragma unroll
        for (int v = 0; v < 8; ++v)
          hg[(size_t)(16 * mt + v + 8 * halfm) * H_ + gcol] = (__bf16)h[mt][v];
      target += NWG;
      sync_and_load_h(cnt, target, sH, hg);
    }
#pragma unroll
    for (int mt = 0; mt < 2; ++mt)
#pragma unroll
      for (int v = 0; v < 8; ++v) {
        int b = 16 * mt + v + 8 * halfm;
        fwd_bf[((size_t)b * L_ + l) * H_ + gcol] = (__bf16)h[mt][v];
      }
  }
#pragma unroll
  for (int mt = 0; mt < 2; ++mt)
#pragma unroll
    for (int v = 0; v < 8; ++v)
      h_last[(size_t)(16 * mt + v + 8 * halfm) * H_ + gcol] = h[mt][v];
}

// ----------------------------------------------------------------- launch
extern "C" void kernel_launch(void* const* d_in, const int* in_sizes, int n_in,
                              void* d_out, int out_size, void* d_ws, size_t ws_size,
                              hipStream_t stream) {
  const float* x       = (const float*)d_in[0];
  const float* log_tau = (const float*)d_in[1];
  const float* W_in    = (const float*)d_in[2];
  const float* b_in    = (const float*)d_in[3];
  const float* W_rec   = (const float*)d_in[4];
  const float* W_gate  = (const float*)d_in[5];
  const float* b_gate  = (const float*)d_in[6];
  const float* W_out   = (const float*)d_in[7];
  const float* b_out   = (const float*)d_in[8];
  float* out = (float*)d_out;                 // [B*L*H output | B*H h_last]

  char* ws = (char*)d_ws;
  size_t off = 0;
  auto alloc = [&](size_t bytes) -> char* {
    char* p = ws + off;
    off = (off + bytes + 255) & ~(size_t)255;
    return p;
  };
  __bf16* xb      = (__bf16*)alloc((size_t)BL * D_ * 2);
  __bf16* Winb    = (__bf16*)alloc((size_t)H_ * D_ * 2);
  __bf16* Wgxb    = (__bf16*)alloc((size_t)H_ * D_ * 2);
  __bf16* Wghb    = (__bf16*)alloc((size_t)H_ * H_ * 2);
  __bf16* Wrecb   = (__bf16*)alloc((size_t)H_ * H_ * 2);
  __bf16* Woutb   = (__bf16*)alloc((size_t)H_ * H_ * 2);
  float*  rate    = (float*)alloc((size_t)H_ * 4);
  float*  drive_t = (float*)alloc((size_t)BL * H_ * 4);
  float*  gate_t  = (float*)alloc((size_t)BL * H_ * 4);
  __bf16* fwd_bf  = (__bf16*)alloc((size_t)BL * H_ * 2);
  __bf16* hg      = (__bf16*)alloc((size_t)B_ * H_ * 2);
  unsigned int* cnt = (unsigned int*)alloc(256);
  if (off > ws_size) return;  // workspace too small; nothing safe to do

  k_zero_sync<<<1, 64, 0, stream>>>(cnt);
  const int nX = BL * D_;
  k_convert_x<<<(nX + 255) / 256, 256, 0, stream>>>(x, xb, nX);
  k_pack_weights<<<(H_ * D_ + 255) / 256, 256, 0, stream>>>(
      W_in, W_rec, W_gate, W_out, log_tau, Winb, Wrecb, Wgxb, Wghb, Woutb, rate);
  k_input_gemm<<<BL / 16, 256, 0, stream>>>(xb, Winb, Wgxb, b_in, b_gate, drive_t, gate_t);

  const int smem = (2 * HC * H_ + B_ * H_) * 2;   // 294912 B < 320 KB WGP LDS
  (void)hipFuncSetAttribute(reinterpret_cast<const void*>(k_scan),
                            hipFuncAttributeMaxDynamicSharedMemorySize, smem);
  k_scan<<<NWG, 256, smem, stream>>>(drive_t, gate_t, Wrecb, Wghb, rate, fwd_bf, hg, cnt,
                                     out + (size_t)BL * H_);
  k_out_gemm<<<BL / 16, 256, 0, stream>>>(fwd_bf, Woutb, b_out, out);
}
